// Bert_BiLSTM_CRF_26276609917485
// MI455X (gfx1250) — matrix operand order repaired and checked by reading, not compiled, not run on hardware
//
#include <hip/hip_runtime.h>
#include <hip/hip_bf16.h>
#include <math.h>
#include <stdint.h>

typedef _Float16 half_t;
typedef __attribute__((ext_vector_type(16))) _Float16 v16h;
typedef __attribute__((ext_vector_type(8)))  _Float16 v8h;
typedef __attribute__((ext_vector_type(8)))  float    v8f;

#define BB   64      // batch
#define TT   256     // seq len
#define DD   768     // input feature dim (also 2H for layer 1)
#define HH   384     // hidden per direction
#define GG   1536    // 4*H gates
#define KK   12      // tagset
#define NEGV (-10000.0f)
#define START_ID 10

// ---------------------------------------------------------------------------
// CDNA5 async global->LDS copy (ASYNCcnt-tracked), 16B per lane.
// VDST = LDS byte offset (low 32 bits of a generic shared pointer),
// VADDR = 64-bit global address.
// ---------------------------------------------------------------------------
__device__ __forceinline__ void async_ld_b128(void* lds_ptr, const void* gptr) {
    unsigned lds_off = (unsigned)(uintptr_t)lds_ptr;
    asm volatile("global_load_async_to_lds_b128 %0, %1, off"
                 :: "v"(lds_off), "v"(gptr) : "memory");
}
__device__ __forceinline__ void wait_async0() {
    asm volatile("s_wait_asynccnt 0" ::: "memory");
}

// ---------------------------------------------------------------------------
// fp32 -> fp16 conversion (weights + activations)
// ---------------------------------------------------------------------------
__global__ void cvt_f32_to_f16(const float* __restrict__ src,
                               half_t* __restrict__ dst, int n) {
    int i = blockIdx.x * blockDim.x + threadIdx.x;
    if (i < n) dst[i] = (half_t)src[i];
}

// ---------------------------------------------------------------------------
// WMMA GEMM:  C[M,N](f16) = A[M,K](f16) @ W[N,K]^T(f16) + bias[N](f32)
// Block tile 64(M) x 128(N), 8 waves (4(M) x 2(N)), K-step 32.
// Double-buffered LDS tiles filled by async global->LDS copies: tile k+32
// streams in while tile k feeds v_wmma. One barrier per K-step.
// M%64==0, N%128==0, K%32==0 assumed (16384, 1536, 768).
// ---------------------------------------------------------------------------
__global__ __launch_bounds__(256) void gemm_f16_wmma(
    const half_t* __restrict__ A, const half_t* __restrict__ W,
    const float* __restrict__ bias, half_t* __restrict__ C,
    int M, int N, int Kd)
{
    __shared__ __align__(16) half_t Asub[2][64][40];   // +8 halves pad/row
    __shared__ __align__(16) half_t Bsub[2][128][40];

    const int tid    = threadIdx.x;
    const int l      = tid & 31;
    const int lane16 = l & 15;
    const int hi     = l >> 4;          // lane-group within wave
    const int w      = tid >> 5;        // wave 0..7
    const int wm     = w >> 1;          // 0..3 (M sub-tile)
    const int wn     = w & 1;           // 0..1 (N sub-tile of 64)
    const int m0     = blockIdx.x * 64;
    const int n0     = blockIdx.y * 128;

    // per-thread staging coordinates (one A chunk, two B chunks of 8 halves)
    const int arow = tid >> 2, ac4 = (tid & 3) * 8;
    const int brow0 = tid >> 2,          bc40 = (tid & 3) * 8;
    const int brow1 = (tid + 256) >> 2,  bc41 = (tid & 3) * 8;

    v8f acc[4];
    #pragma unroll
    for (int nt = 0; nt < 4; ++nt)
        #pragma unroll
        for (int r = 0; r < 8; ++r) acc[nt][r] = 0.f;

    auto stage = [&](int buf, int kk) {
        async_ld_b128(&Asub[buf][arow][ac4],
                      A + (size_t)(m0 + arow) * Kd + kk + ac4);
        async_ld_b128(&Bsub[buf][brow0][bc40],
                      W + (size_t)(n0 + brow0) * Kd + kk + bc40);
        async_ld_b128(&Bsub[buf][brow1][bc41],
                      W + (size_t)(n0 + brow1) * Kd + kk + bc41);
    };

    stage(0, 0);
    wait_async0();
    __syncthreads();

    int cur = 0;
    for (int kk = 0; kk < Kd; kk += 32) {
        if (kk + 32 < Kd) stage(cur ^ 1, kk + 32);   // overlap next tile

        // A fragment: lanes 0-15 -> K 0..15, lanes 16-31 -> K 16..31
        const int kbA = hi ? 16 : 0;
        v8h alo = *(const v8h*)&Asub[cur][wm * 16 + lane16][kbA];
        v8h ahi = *(const v8h*)&Asub[cur][wm * 16 + lane16][kbA + 8];
        v16h a;
        #pragma unroll
        for (int j = 0; j < 8; ++j) { a[j] = alo[j]; a[8 + j] = ahi[j]; }

        // B fragment: lanes 0-15 -> K 0..15, lanes 16-31 -> K 16..31
        const int kbB = hi ? 16 : 0;
        #pragma unroll
        for (int nt = 0; nt < 4; ++nt) {
            const half_t* bp = &Bsub[cur][wn * 64 + nt * 16 + lane16][kbB];
            v8h blo = *(const v8h*)bp;
            v8h bhi = *(const v8h*)(bp + 8);
            v16h b;
            #pragma unroll
            for (int j = 0; j < 8; ++j) { b[j] = blo[j]; b[8 + j] = bhi[j]; }
            acc[nt] = __builtin_amdgcn_wmma_f32_16x16x32_f16(
                false, a, false, b, (short)0, acc[nt], false, false);
        }

        wait_async0();      // my async writes to buf^1 are in LDS
        __syncthreads();    // everyone done reading buf & writing buf^1
        cur ^= 1;
    }

    #pragma unroll
    for (int nt = 0; nt < 4; ++nt)
        #pragma unroll
        for (int r = 0; r < 8; ++r) {
            int m = m0 + wm * 16 + r + hi * 8;
            int n = n0 + wn * 64 + nt * 16 + lane16;
            C[(size_t)m * N + n] = (half_t)(acc[nt][r] + bias[n]);
        }
}

// ---------------------------------------------------------------------------
// BiLSTM scan (one layer, both directions in one grid).
// grid = 8 blocks: blockIdx.x>>2 = direction, &3 = batch group of 16.
// Each block: 16 batch rows, full T loop. Per step: h[16x384](LDS,f16) @
// Whh^T via WMMA (wave owns 48 H-columns across all 4 gates = 12 tiles),
// gates fused, c kept in registers in C-fragment layout.
// ---------------------------------------------------------------------------
__global__ __launch_bounds__(256) void lstm_scan(
    const half_t* __restrict__ pre_f, const half_t* __restrict__ pre_b,
    const half_t* __restrict__ whh_f, const half_t* __restrict__ whh_b,
    half_t* __restrict__ out16, int outStride)
{
    __shared__ __align__(16) half_t hlds[16][392];   // 384 + 8 pad halves

    const int tid    = threadIdx.x;
    const int l      = tid & 31;
    const int lane16 = l & 15;
    const int hi     = l >> 4;
    const int w      = tid >> 5;                 // wave 0..7
    const int dir    = blockIdx.x >> 2;          // 0 fwd, 1 bwd
    const int b0     = (blockIdx.x & 3) * 16;

    const half_t* pre = dir ? pre_b : pre_f;
    const half_t* whh = dir ? whh_b : whh_f;
    const int colofs  = dir ? HH : 0;
    const int j0      = w * 48;                  // this wave's H-column base

    for (int i = tid; i < 16 * 392; i += 256) ((half_t*)hlds)[i] = (half_t)0.f;

    float c[3][8];
    #pragma unroll
    for (int s = 0; s < 3; ++s)
        #pragma unroll
        for (int r = 0; r < 8; ++r) c[s][r] = 0.f;

    __syncthreads();

    for (int t = 0; t < TT; ++t) {
        const int tt = dir ? (TT - 1 - t) : t;

        v8f acc[4][3];
        #pragma unroll
        for (int g = 0; g < 4; ++g)
            #pragma unroll
            for (int s = 0; s < 3; ++s)
                #pragma unroll
                for (int r = 0; r < 8; ++r) acc[g][s][r] = 0.f;

        #pragma unroll 1
        for (int k0 = 0; k0 < HH; k0 += 32) {
            const int kbA = hi ? 16 : 0;
            v8h alo = *(const v8h*)&hlds[lane16][k0 + kbA];
            v8h ahi = *(const v8h*)&hlds[lane16][k0 + kbA + 8];
            v16h a;
            #pragma unroll
            for (int j = 0; j < 8; ++j) { a[j] = alo[j]; a[8 + j] = ahi[j]; }

            const int kbB = hi ? 16 : 0;
            #pragma unroll
            for (int g = 0; g < 4; ++g)
                #pragma unroll
                for (int s = 0; s < 3; ++s) {
                    const int nn = g * HH + j0 + s * 16 + lane16;
                    const half_t* bp = whh + (size_t)nn * HH + k0 + kbB;
                    v8h blo = *(const v8h*)bp;
                    v8h bhi = *(const v8h*)(bp + 8);
                    v16h b;
                    #pragma unroll
                    for (int j = 0; j < 8; ++j) { b[j] = blo[j]; b[8 + j] = bhi[j]; }
                    acc[g][s] = __builtin_amdgcn_wmma_f32_16x16x32_f16(
                        false, a, false, b, (short)0, acc[g][s], false, false);
                }
        }
        __syncthreads();   // all reads of hlds done before overwrite

        #pragma unroll
        for (int s = 0; s < 3; ++s)
            #pragma unroll
            for (int r = 0; r < 8; ++r) {
                const int m = r + hi * 8;
                const int j = j0 + s * 16 + lane16;
                const size_t rowbase = ((size_t)(b0 + m) * TT + tt) * GG;
                float gi = acc[0][s][r] + (float)pre[rowbase + 0 * HH + j];
                float gf = acc[1][s][r] + (float)pre[rowbase + 1 * HH + j];
                float gg = acc[2][s][r] + (float)pre[rowbase + 2 * HH + j];
                float go = acc[3][s][r] + (float)pre[rowbase + 3 * HH + j];
                float si = 1.f / (1.f + __expf(-gi));
                float sf = 1.f / (1.f + __expf(-gf));
                float tg = 2.f / (1.f + __expf(-2.f * gg)) - 1.f;
                float so = 1.f / (1.f + __expf(-go));
                float cn = sf * c[s][r] + si * tg;
                c[s][r] = cn;
                float tc = 2.f / (1.f + __expf(-2.f * cn)) - 1.f;
                float hv = so * tc;
                hlds[m][j] = (half_t)hv;
                out16[((size_t)(b0 + m) * TT + tt) * (size_t)outStride + colofs + j] =
                    (half_t)hv;
            }
        __syncthreads();   // hlds updated before next step's WMMA reads
    }
}

// ---------------------------------------------------------------------------
// feats[m,k] = h2[m,:768] . fcW[k,:768] + fcb[k]   (K=12, trivial)
// ---------------------------------------------------------------------------
__global__ void feats_kernel(const half_t* __restrict__ h2,
                             const float* __restrict__ fcW,
                             const float* __restrict__ fcb,
                             float* __restrict__ feats, int M) {
    int idx = blockIdx.x * blockDim.x + threadIdx.x;
    if (idx >= M * KK) return;
    int m = idx / KK, k = idx % KK;
    const half_t* hr = h2 + (size_t)m * DD;
    const float*  wr = fcW + (size_t)k * DD;
    float s = fcb[k];
    #pragma unroll 4
    for (int j = 0; j < DD; ++j) s += (float)hr[j] * wr[j];
    feats[idx] = s;
}

// ---------------------------------------------------------------------------
// Viterbi: one thread per batch element (B=64).
// out[0..63] = best score; out[64 + b*256 + t] = (float)tag.
// ---------------------------------------------------------------------------
__global__ void viterbi_kernel(const float* __restrict__ feats,
                               const float* __restrict__ trans,
                               float* __restrict__ out,
                               int* __restrict__ psi) {
    int b = blockIdx.x * blockDim.x + threadIdx.x;
    if (b >= BB) return;
    float delta[KK];
    #pragma unroll
    for (int j = 0; j < KK; ++j) delta[j] = NEGV;
    delta[START_ID] = 0.f;
    const float* fb = feats + (size_t)b * TT * KK;
    int* pb = psi + (size_t)b * (TT - 1) * KK;
    for (int t = 1; t < TT; ++t) {
        float nd[KK];
        #pragma unroll
        for (int j = 0; j < KK; ++j) {
            float best = trans[j * KK] + delta[0];
            int bi = 0;
            #pragma unroll
            for (int i = 1; i < KK; ++i) {
                float v = trans[j * KK + i] + delta[i];
                if (v > best) { best = v; bi = i; }
            }
            nd[j] = best + fb[t * KK + j];
            pb[(t - 1) * KK + j] = bi;
        }
        #pragma unroll
        for (int j = 0; j < KK; ++j) delta[j] = nd[j];
    }
    float best = delta[0]; int last = 0;
    #pragma unroll
    for (int j = 1; j < KK; ++j)
        if (delta[j] > best) { best = delta[j]; last = j; }
    out[b] = best;
    float* path = out + BB + (size_t)b * TT;
    path[TT - 1] = (float)last;
    int idx = last;
    for (int t = TT - 2; t >= 0; --t) {
        idx = pb[t * KK + idx];
        path[t] = (float)idx;
    }
}

// ---------------------------------------------------------------------------
// Host launcher
// ---------------------------------------------------------------------------
extern "C" void kernel_launch(void* const* d_in, const int* in_sizes, int n_in,
                              void* d_out, int out_size, void* d_ws, size_t ws_size,
                              hipStream_t stream) {
    const float* embeds   = (const float*)d_in[0];
    const float* Wih[4]   = { (const float*)d_in[1], (const float*)d_in[4],
                              (const float*)d_in[7], (const float*)d_in[10] }; // l0f,l0b,l1f,l1b
    const float* Whh[4]   = { (const float*)d_in[2], (const float*)d_in[5],
                              (const float*)d_in[8], (const float*)d_in[11] };
    const float* bias[4]  = { (const float*)d_in[3], (const float*)d_in[6],
                              (const float*)d_in[9], (const float*)d_in[12] };
    const float* fcW      = (const float*)d_in[13];
    const float* fcb      = (const float*)d_in[14];
    const float* trans    = (const float*)d_in[15];

    const int M = BB * TT;                 // 16384 rows
    char* ws = (char*)d_ws;
    size_t o = 0;
    auto take = [&](size_t bytes) { char* p = ws + o; o = (o + bytes + 255) & ~(size_t)255; return p; };

    half_t* embA   = (half_t*)take((size_t)M * DD * 2);          // 25.2 MB
    half_t* wih16[4];
    for (int i = 0; i < 4; ++i) wih16[i] = (half_t*)take((size_t)GG * DD * 2);
    half_t* whh16[4];
    for (int i = 0; i < 4; ++i) whh16[i] = (half_t*)take((size_t)GG * HH * 2);
    half_t* pre_f  = (half_t*)take((size_t)M * GG * 2);          // 50.3 MB
    half_t* pre_b  = (half_t*)take((size_t)M * GG * 2);          // 50.3 MB
    half_t* h1     = (half_t*)take((size_t)M * DD * 2);          // 25.2 MB
    half_t* h2     = (half_t*)take((size_t)M * DD * 2);          // 25.2 MB
    float*  featsb = (float*) take((size_t)M * KK * 4);
    int*    psib   = (int*)   take((size_t)BB * (TT - 1) * KK * 4);

    // 1) fp32 -> fp16 conversions
    {
        int n = M * DD;
        cvt_f32_to_f16<<<(n + 255) / 256, 256, 0, stream>>>(embeds, embA, n);
        n = GG * DD;
        for (int i = 0; i < 4; ++i)
            cvt_f32_to_f16<<<(n + 255) / 256, 256, 0, stream>>>(Wih[i], wih16[i], n);
        n = GG * HH;
        for (int i = 0; i < 4; ++i)
            cvt_f32_to_f16<<<(n + 255) / 256, 256, 0, stream>>>(Whh[i], whh16[i], n);
    }

    dim3 gemmGrid(M / 64, GG / 128);

    // 2) layer-0 input projections (pre = x @ Wih^T + b)
    gemm_f16_wmma<<<gemmGrid, 256, 0, stream>>>(embA, wih16[0], bias[0], pre_f, M, GG, DD);
    gemm_f16_wmma<<<gemmGrid, 256, 0, stream>>>(embA, wih16[1], bias[1], pre_b, M, GG, DD);

    // 3) layer-0 scan (fwd+bwd concurrently) -> h1 (f16, [M,768])
    lstm_scan<<<8, 256, 0, stream>>>(pre_f, pre_b, whh16[0], whh16[1], h1, DD);

    // 4) layer-1 input projections (reuse pre buffers)
    gemm_f16_wmma<<<gemmGrid, 256, 0, stream>>>(h1, wih16[2], bias[2], pre_f, M, GG, DD);
    gemm_f16_wmma<<<gemmGrid, 256, 0, stream>>>(h1, wih16[3], bias[3], pre_b, M, GG, DD);

    // 5) layer-1 scan -> h2 (f16)
    lstm_scan<<<8, 256, 0, stream>>>(pre_f, pre_b, whh16[2], whh16[3], h2, DD);

    // 6) FC head
    {
        int n = M * KK;
        feats_kernel<<<(n + 255) / 256, 256, 0, stream>>>(h2, fcW, fcb, featsb, M);
    }

    // 7) Viterbi decode -> d_out (score[64] ++ path[64*256] as float)
    viterbi_kernel<<<1, 64, 0, stream>>>(featsb, trans, (float*)d_out, psib);
}
